// CapsEEGNet_41068477285255
// MI455X (gfx1250) — compile-verified, matrix operand order checked by reading, not run on hardware
//
#include <hip/hip_runtime.h>
#include <hip/hip_bf16.h>
#include <math.h>

typedef __attribute__((ext_vector_type(16))) __bf16 v16bf;
typedef __attribute__((ext_vector_type(8)))  __bf16 v8bf;
typedef __attribute__((ext_vector_type(8)))  float  v8f;
typedef __attribute__((ext_vector_type(4)))  unsigned int v4u;
typedef __attribute__((ext_vector_type(8)))  int v8i;
typedef __attribute__((ext_vector_type(4)))  int v4i;

#define SDIM 128
#define CHANS 32

// ---------------- workspace layout (bytes) ----------------
static constexpr size_t OFF_H    = 0;                      // B*16*128 f32   = 2 MB
static constexpr size_t OFF_WEFF = 2097152;                // 24576 bf16     = 48 KB (A-fragments)
static constexpr size_t OFF_BEFF = OFF_WEFF + 49152;       // 256 f32
static constexpr size_t OFF_PC2  = OFF_BEFF + 1280;        // B*256*128 f32  = 33.5 MB
static constexpr size_t OFF_UHAT = OFF_PC2 + (size_t)256*256*128*4; // B*4096*64 bf16 = 134 MB

__device__ __forceinline__ float elu1(float x) { return x > 0.f ? x : expm1f(x); }

// -------- Tensor Data Mover: 1-row f32 tile, global -> LDS (ISA 08 §8) ----
// D# group0: [1:0]=count=1, [63:32]=lds_addr, [120:64]=global_addr, [127:126]=type(2)
// D# group1: [17:16]=data_size(2 -> 4B), [79:48]=tensor_dim0, [111:80]=tensor_dim1,
//            [127:112]=tile_dim0, [143:128]=tile_dim1, [207:160]=tensor_dim0_stride
__device__ __forceinline__ void tdm_load_f32(const float* gsrc, void* lds_dst,
                                             unsigned nelem /* < 65536 */) {
  unsigned ldsa = (unsigned)(unsigned long long)lds_dst;       // LDS addr = low 32b
  unsigned long long ga = (unsigned long long)gsrc;
  v4u g0 = { 1u,
             ldsa,
             (unsigned)ga,
             (unsigned)((ga >> 32) & 0x01FFFFFFu) | 0x80000000u };
  v8i g1 = { (int)(2u << 16),                       // data_size = 4B
             (int)((nelem & 0xFFFFu) << 16),        // tensor_dim0 lo16
             (int)((nelem >> 16) | (1u << 16)),     // tensor_dim0 hi16, tensor_dim1 = 1
             (int)(nelem << 16),                    // tile_dim0 = nelem
             1,                                     // tile_dim1 = 1
             (int)nelem,                            // tensor_dim0_stride lo32
             0, 0 };
  v4i z4 = { 0, 0, 0, 0 };
  v8i z8 = { 0, 0, 0, 0, 0, 0, 0, 0 };
  __builtin_amdgcn_tensor_load_to_lds(g0, g1, z4, z4, z8, 0);
}

// =====================================================================
// Kernel 1: conv1(k=64,'same') + BN1 + ELU + constrained depthwise conv
//           over Chans + BN2 + ELU  ->  h (B,16,128) f32
// =====================================================================
__global__ __launch_bounds__(128) void k_eegnet(
    const float* __restrict__ x,  const float* __restrict__ w1,
    const float* __restrict__ g1, const float* __restrict__ b1,
    const float* __restrict__ m1, const float* __restrict__ v1,
    const float* __restrict__ wdw,
    const float* __restrict__ g2, const float* __restrict__ b2,
    const float* __restrict__ m2, const float* __restrict__ v2,
    float* __restrict__ hout)
{
  __shared__ float xsh[CHANS][SDIM];
  __shared__ float w1sh[8][64];
  __shared__ float wdwsh[16][CHANS];
  __shared__ float a1[8], c1[8], a2[16], c2[16];
  const int b = blockIdx.x, tid = threadIdx.x;
  const float* xb = x + (size_t)b * CHANS * SDIM;

  // TDM: DMA x[b] (16 KB) into LDS from wave 0 (scalar-branch guard; TDM ignores EXEC)
  const bool wave0 = __builtin_amdgcn_readfirstlane(tid >> 5) == 0;
  if (wave0) tdm_load_f32(xb, &xsh[0][0], CHANS * SDIM);

  for (int i = tid; i < 512; i += 128) w1sh[i >> 6][i & 63] = w1[i];
  if (tid < 16) {
    float nrm = 0.f;
    for (int c = 0; c < CHANS; ++c) { float w = wdw[tid * CHANS + c]; nrm = fmaf(w, w, nrm); }
    nrm = sqrtf(nrm);
    float sc = nrm > 1.0f ? 1.0f / (nrm + 1e-7f) : 1.0f;
    for (int c = 0; c < CHANS; ++c) wdwsh[tid][c] = wdw[tid * CHANS + c] * sc;
    float inv = g2[tid] * rsqrtf(v2[tid] + 1e-5f);
    a2[tid] = inv; c2[tid] = b2[tid] - m2[tid] * inv;
  }
  if (tid < 8) {
    float inv = g1[tid] * rsqrtf(v1[tid] + 1e-5f);
    a1[tid] = inv; c1[tid] = b1[tid] - m1[tid] * inv;
  }
  if (wave0) __builtin_amdgcn_s_wait_tensorcnt(0);
  __syncthreads();

  const int s = tid;
  float hacc[16];
  #pragma unroll
  for (int o = 0; o < 16; ++o) hacc[o] = 0.f;
  for (int c = 0; c < CHANS; ++c) {
    float facc[8] = {0.f, 0.f, 0.f, 0.f, 0.f, 0.f, 0.f, 0.f};
    for (int t = 0; t < 64; ++t) {
      int idx = s + t - 31;
      float xv = (idx >= 0 && idx < SDIM) ? xsh[c][idx] : 0.f;
      #pragma unroll
      for (int f = 0; f < 8; ++f) facc[f] = fmaf(xv, w1sh[f][t], facc[f]);
    }
    #pragma unroll
    for (int f = 0; f < 8; ++f) {
      float e = elu1(facc[f] * a1[f] + c1[f]);
      hacc[2 * f]     = fmaf(e, wdwsh[2 * f][c],     hacc[2 * f]);
      hacc[2 * f + 1] = fmaf(e, wdwsh[2 * f + 1][c], hacc[2 * f + 1]);
    }
  }
  float* hb = hout + (size_t)b * 16 * SDIM;
  #pragma unroll
  for (int o = 0; o < 16; ++o) hb[o * SDIM + s] = elu1(hacc[o] * a2[o] + c2[o]);
}

// =====================================================================
// Kernel 2: fold pc2 into pc:  Weff[o2,(cin,t)] = sum_o pc2w[o2,16+o]*pcw[o,cin,t]
//           (+ pc2w[o2,cin] at t==2),  beff = pc2b + pc2w[:,16:]@pcb.
//           Weff written as WMMA A-fragments: [mt][ks][lane][e] bf16.
// =====================================================================
__global__ __launch_bounds__(256) void k_prep(
    const float* __restrict__ pcw,  const float* __restrict__ pcb,
    const float* __restrict__ pc2w, const float* __restrict__ pc2b,
    __bf16* __restrict__ weff, float* __restrict__ beff)
{
  const int blk = blockIdx.x, tid = threadIdx.x;
  if (blk == 96) {
    float acc = pc2b[tid];
    for (int o = 0; o < 256; ++o) acc = fmaf(pc2w[tid * 272 + 16 + o], pcb[o], acc);
    beff[tid] = acc;
    return;
  }
  const int gid = blk * 256 + tid;
  const int e = gid & 15, lane = (gid >> 4) & 31, r = gid >> 9;
  const int ks = r % 3, mt = r / 3;
  const int lo = lane & 15, hi = lane >> 4;
  const int kk = ks * 32 + (e & 7) + 8 * hi + ((e >> 3) << 4);  // 0..95
  const int o2 = mt * 16 + lo;
  const int cin = kk / 6, t = kk % 6;
  float acc = (t == 2) ? pc2w[o2 * 272 + cin] : 0.f;
  for (int o = 0; o < 256; ++o) acc = fmaf(pc2w[o2 * 272 + 16 + o], pcw[(o * 16 + cin) * 6 + t], acc);
  weff[gid] = (__bf16)acc;
}

// =====================================================================
// Kernel 3: per-b WMMA GEMM  out(256x128) = Weff(256x96) @ im2col(h)(96x128) + beff
//           one block per b, 8 waves, 16 output tiles per wave, K = 3 x 32.
//           h staged to LDS via TDM.
// =====================================================================
__global__ __launch_bounds__(256) void k_gemm(
    const float* __restrict__ h, const v16bf* __restrict__ weff,
    const float* __restrict__ beff, float* __restrict__ out)
{
  __shared__ float hsh[16][SDIM];
  __shared__ __align__(16) __bf16 BmT[SDIM][96];   // [s][cin*6+t]
  const int b = blockIdx.x, tid = threadIdx.x;
  const float* hb = h + (size_t)b * 16 * SDIM;

  if (__builtin_amdgcn_readfirstlane(tid >> 5) == 0) {
    tdm_load_f32(hb, &hsh[0][0], 16 * SDIM);       // 8 KB DMA, TENSORcnt
    __builtin_amdgcn_s_wait_tensorcnt(0);
  }
  __syncthreads();

  for (int i = tid; i < SDIM * 96; i += 256) {
    int s = i / 96, kk = i % 96;
    int cin = kk / 6, t = kk % 6;
    int sp = s + t - 2;
    BmT[s][kk] = (sp >= 0 && sp < SDIM) ? (__bf16)hsh[cin][sp] : (__bf16)0.f;
  }
  __syncthreads();
  const int wave = tid >> 5, lane = tid & 31;
  const int lo = lane & 15, hi = lane >> 4;
  float* ob = out + (size_t)b * 256 * SDIM;
  for (int ti = 0; ti < 16; ++ti) {
    const int tile = wave * 16 + ti;
    const int mt = tile >> 3, nt = tile & 7;
    v8f acc = {0.f, 0.f, 0.f, 0.f, 0.f, 0.f, 0.f, 0.f};
    #pragma unroll
    for (int ks = 0; ks < 3; ++ks) {
      v16bf afrag = weff[(mt * 3 + ks) * 32 + lane];          // 32B coalesced
      const int cb = ks * 32 + 8 * hi;
      v8bf blo = *(const v8bf*)&BmT[nt * 16 + lo][cb];
      v8bf bhi = *(const v8bf*)&BmT[nt * 16 + lo][cb + 16];
      v16bf bfrag;
      #pragma unroll
      for (int q = 0; q < 8; ++q) { bfrag[q] = blo[q]; bfrag[q + 8] = bhi[q]; }
      acc = __builtin_amdgcn_wmma_f32_16x16x32_bf16(
          false, afrag, false, bfrag, (short)0, acc, false, false);
    }
    #pragma unroll
    for (int v = 0; v < 8; ++v) {
      const int o2 = mt * 16 + v + 8 * hi;
      ob[o2 * SDIM + nt * 16 + lo] = acc[v] + beff[o2];
    }
  }
}

// =====================================================================
// Kernel 4: per-b squash -> u, u_hat (bf16 to L2-resident ws), 3-round
//           dynamic routing with b_kn held in VGPRs, head softmax.
// =====================================================================
__global__ __launch_bounds__(256) void k_route(
    const float* __restrict__ pc2out, const float* __restrict__ emW,
    const float* __restrict__ fcw,    const float* __restrict__ fcb,
    __bf16* __restrict__ uhat, float* __restrict__ dout)
{
  __shared__ float sred[64];
  __shared__ float vsh[64];
  __shared__ float lsh[4];
  const int b = blockIdx.x, tid = threadIdx.x;
  const float* ob = pc2out + (size_t)b * 256 * SDIM;
  __bf16* ub = uhat + (size_t)b * 4096 * 64;
  float accs[64];   // partial s[k][d]
  float breg[64];   // routing logits b_kn for owned n's: [j][k]
  #pragma unroll
  for (int q = 0; q < 64; ++q) { accs[q] = 0.f; breg[q] = 0.f; }

  // ---- phase A: u = squash(out), u_hat = W @ u, accumulate iter-0 s (c = 1/4)
  for (int j = 0; j < 16; ++j) {
    const int n = tid + j * 256;
    if (j < 15) __builtin_prefetch(emW + ((size_t)(n + 256)) * 128, 0, 0);
    const float* base = ob + (n >> 4) * SDIM + (n & 15) * 8;
    float un[8]; float sq = 0.f;
    #pragma unroll
    for (int i = 0; i < 8; ++i) { un[i] = base[i]; sq = fmaf(un[i], un[i], sq); }
    sq += 8.f * 1e-7f;
    const float f = sq / ((1.f + sq) * sqrtf(sq));
    #pragma unroll
    for (int i = 0; i < 8; ++i) un[i] *= f;
    __bf16* uw = ub + (size_t)n * 64;
    for (int k = 0; k < 4; ++k) {
      const float* W = emW + ((size_t)(k * 4096 + n)) * 128;  // [d][i]
      #pragma unroll
      for (int d = 0; d < 16; ++d) {
        float uh = 0.f;
        #pragma unroll
        for (int i = 0; i < 8; ++i) uh = fmaf(W[d * 8 + i], un[i], uh);
        uw[k * 16 + d] = (__bf16)uh;
        accs[k * 16 + d] = fmaf(0.25f, uh, accs[k * 16 + d]);
      }
    }
  }

  // ---- routing iterations
  for (int iter = 0; iter < 3; ++iter) {
    if (tid < 64) sred[tid] = 0.f;
    __syncthreads();
    #pragma unroll
    for (int q = 0; q < 64; ++q) atomicAdd(&sred[q], accs[q]);
    __syncthreads();
    if (tid < 4) {
      float sq = 0.f;
      for (int d = 0; d < 16; ++d) { float s = sred[tid * 16 + d]; sq = fmaf(s, s, sq); }
      sq += 16.f * 1e-7f;
      const float f = sq / ((1.f + sq) * sqrtf(sq));
      for (int d = 0; d < 16; ++d) vsh[tid * 16 + d] = sred[tid * 16 + d] * f;
    }
    __syncthreads();
    if (iter == 2) break;
    #pragma unroll
    for (int q = 0; q < 64; ++q) accs[q] = 0.f;
    for (int j = 0; j < 16; ++j) {
      const int n = tid + j * 256;
      const __bf16* uw = ub + (size_t)n * 64;
      float uh[64];
      #pragma unroll
      for (int q = 0; q < 64; ++q) uh[q] = (float)uw[q];
      float bk[4];
      #pragma unroll
      for (int k = 0; k < 4; ++k) {
        float d0 = 0.f;
        #pragma unroll
        for (int d = 0; d < 16; ++d) d0 = fmaf(uh[k * 16 + d], vsh[k * 16 + d], d0);
        bk[k] = breg[j * 4 + k] + d0;
        breg[j * 4 + k] = bk[k];
      }
      const float m = fmaxf(fmaxf(bk[0], bk[1]), fmaxf(bk[2], bk[3]));
      float ex[4], es = 0.f;
      #pragma unroll
      for (int k = 0; k < 4; ++k) { ex[k] = __expf(bk[k] - m); es += ex[k]; }
      const float inv = 1.f / es;
      #pragma unroll
      for (int k = 0; k < 4; ++k) {
        const float c = ex[k] * inv;
        #pragma unroll
        for (int d = 0; d < 16; ++d) accs[k * 16 + d] = fmaf(c, uh[k * 16 + d], accs[k * 16 + d]);
      }
    }
    __syncthreads();
  }

  // ---- head: Linear(16,1) + softmax over K=4
  if (tid < 4) {
    float l = fcb[0];
    for (int d = 0; d < 16; ++d) l = fmaf(vsh[tid * 16 + d], fcw[d], l);
    lsh[tid] = l;
  }
  __syncthreads();
  if (tid < 4) {
    const float m = fmaxf(fmaxf(lsh[0], lsh[1]), fmaxf(lsh[2], lsh[3]));
    const float es = __expf(lsh[0] - m) + __expf(lsh[1] - m) +
                     __expf(lsh[2] - m) + __expf(lsh[3] - m);
    dout[b * 4 + tid] = __expf(lsh[tid] - m) / es;
  }
}

// =====================================================================
extern "C" void kernel_launch(void* const* d_in, const int* in_sizes, int n_in,
                              void* d_out, int out_size, void* d_ws, size_t ws_size,
                              hipStream_t stream) {
  const float* x     = (const float*)d_in[0];
  const float* w1    = (const float*)d_in[1];
  const float* g1    = (const float*)d_in[2];
  const float* b1    = (const float*)d_in[3];
  const float* m1    = (const float*)d_in[4];
  const float* v1    = (const float*)d_in[5];
  const float* wdw   = (const float*)d_in[6];
  const float* g2    = (const float*)d_in[7];
  const float* b2    = (const float*)d_in[8];
  const float* m2    = (const float*)d_in[9];
  const float* v2    = (const float*)d_in[10];
  const float* pcw   = (const float*)d_in[11];
  const float* pcb   = (const float*)d_in[12];
  const float* pc2w  = (const float*)d_in[13];
  const float* pc2b  = (const float*)d_in[14];
  const float* emW   = (const float*)d_in[15];
  const float* fcw   = (const float*)d_in[16];
  const float* fcb   = (const float*)d_in[17];

  const int B = in_sizes[0] / (CHANS * SDIM);   // 256

  char* ws = (char*)d_ws;
  float*  h      = (float*)(ws + OFF_H);
  __bf16* weff   = (__bf16*)(ws + OFF_WEFF);
  float*  beff   = (float*)(ws + OFF_BEFF);
  float*  pc2out = (float*)(ws + OFF_PC2);
  __bf16* uhat   = (__bf16*)(ws + OFF_UHAT);

  k_eegnet<<<B, 128, 0, stream>>>(x, w1, g1, b1, m1, v1, wdw, g2, b2, m2, v2, h);
  k_prep  <<<97, 256, 0, stream>>>(pcw, pcb, pc2w, pc2b, weff, beff);
  k_gemm  <<<B, 256, 0, stream>>>(h, (const v16bf*)weff, beff, pc2out);
  k_route <<<B, 256, 0, stream>>>(pc2out, emW, fcw, fcb, uhat, (float*)d_out);
}
